// GNNEncoder_15616501088448
// MI455X (gfx1250) — matrix operand order repaired and checked by reading, not compiled, not run on hardware
//
#include <hip/hip_runtime.h>
#include <hip/hip_bf16.h>

#define NN 50000
#define EE 625000
#define HH 128
#define LL 4
#define GG 2000

typedef float v8f     __attribute__((ext_vector_type(8)));
typedef __bf16 bf16x16 __attribute__((ext_vector_type(16)));
typedef __bf16 bf16x8  __attribute__((ext_vector_type(8)));

#define ZS 136   // 128 bf16 + 8 pad  (272B row -> 4-bank rotation)
#define US 264   // 256 bf16 + 8 pad
#define YS 132   // 128 f32 + 4 pad

// ---------------------------------------------------------------------------
// Convert W1 [L][128][256] and W2 [L][256][128] (f32, row-major) into
// transposed bf16: W1t[l][n(256)][k(128)], W2t[l][n(128)][k(256)].
// ---------------------------------------------------------------------------
__global__ void k_wconv(const float* __restrict__ W1, const float* __restrict__ W2,
                        __bf16* __restrict__ W1t, __bf16* __restrict__ W2t) {
    int t = blockIdx.x * blockDim.x + threadIdx.x;   // 0 .. L*H*2H-1 = 131071
    if (t >= LL * HH * 2 * HH) return;
    int l = t / (HH * 2 * HH);
    int rem = t % (HH * 2 * HH);
    // W1 element: [l][k=rem/256][n=rem%256]
    int k1 = rem / (2 * HH), n1 = rem % (2 * HH);
    W1t[((size_t)l * 2 * HH + n1) * HH + k1] = (__bf16)W1[t];
    // W2 element: [l][k=rem/128][n=rem%128]
    int k2 = rem / HH, n2 = rem % HH;
    W2t[((size_t)l * HH + n2) * 2 * HH + k2] = (__bf16)W2[t];
}

// ---------------------------------------------------------------------------
// Node encoder: h[n][c] = sum_f x[n][f]*Wn[f][c] + bn[c].  One wave per node.
// ---------------------------------------------------------------------------
__global__ void k_encode(const float* __restrict__ x, const float* __restrict__ Wn,
                         const float* __restrict__ bn, float* __restrict__ h) {
    __shared__ float sW[9 * HH];
    __shared__ float sb[HH];
    for (int i = threadIdx.x; i < 9 * HH; i += 256) sW[i] = Wn[i];
    if (threadIdx.x < HH) sb[threadIdx.x] = bn[threadIdx.x];
    __syncthreads();
    int wave = threadIdx.x >> 5, lane = threadIdx.x & 31;
    int n = blockIdx.x * 8 + wave;
    if (n >= NN) return;
    int c = lane * 4;
    float a0 = sb[c], a1 = sb[c + 1], a2 = sb[c + 2], a3 = sb[c + 3];
    for (int f = 0; f < 9; ++f) {
        float xv = x[(size_t)n * 9 + f];
        a0 += xv * sW[f * HH + c];
        a1 += xv * sW[f * HH + c + 1];
        a2 += xv * sW[f * HH + c + 2];
        a3 += xv * sW[f * HH + c + 3];
    }
    float4 o; o.x = a0; o.y = a1; o.z = a2; o.w = a3;
    *(float4*)(h + (size_t)n * HH + c) = o;
}

// ---------------------------------------------------------------------------
// Zero a float buffer (n multiple of 4).
// ---------------------------------------------------------------------------
__global__ void k_zero(float* __restrict__ p, int n) {
    int i = (blockIdx.x * blockDim.x + threadIdx.x) * 4;
    if (i < n) {
        float4 z; z.x = 0.f; z.y = 0.f; z.z = 0.f; z.w = 0.f;
        *(float4*)(p + i) = z;
    }
}

// ---------------------------------------------------------------------------
// Fused edge-encode + message + scatter-aggregate.
// msg = relu(h[src] + edge_attr@We + be); agg[dst] += msg (f32 atomics).
// One wave per edge; lane covers 4 columns.
// ---------------------------------------------------------------------------
__global__ void k_msg_agg(const float* __restrict__ h, const int* __restrict__ ei,
                          const float* __restrict__ eattr, const float* __restrict__ We,
                          const float* __restrict__ be, float* __restrict__ agg) {
    __shared__ float sWe[4 * HH];
    __shared__ float sbe[HH];
    for (int i = threadIdx.x; i < 4 * HH; i += 256) sWe[i] = We[i];
    if (threadIdx.x < HH) sbe[threadIdx.x] = be[threadIdx.x];
    __syncthreads();
    int wave = threadIdx.x >> 5, lane = threadIdx.x & 31;
    int e = blockIdx.x * 8 + wave;
    if (e >= EE) return;
    int src = ei[e];
    int dst = ei[EE + e];
    float e0 = eattr[(size_t)e * 4 + 0];
    float e1 = eattr[(size_t)e * 4 + 1];
    float e2 = eattr[(size_t)e * 4 + 2];
    float e3 = eattr[(size_t)e * 4 + 3];
    int c = lane * 4;
    float4 hv = *(const float4*)(h + (size_t)src * HH + c);
    float* ap = agg + (size_t)dst * HH + c;
    float m0 = hv.x + e0 * sWe[c]      + e1 * sWe[HH + c]     + e2 * sWe[2*HH + c]     + e3 * sWe[3*HH + c]     + sbe[c];
    float m1 = hv.y + e0 * sWe[c + 1]  + e1 * sWe[HH + c + 1] + e2 * sWe[2*HH + c + 1] + e3 * sWe[3*HH + c + 1] + sbe[c + 1];
    float m2 = hv.z + e0 * sWe[c + 2]  + e1 * sWe[HH + c + 2] + e2 * sWe[2*HH + c + 2] + e3 * sWe[3*HH + c + 2] + sbe[c + 2];
    float m3 = hv.w + e0 * sWe[c + 3]  + e1 * sWe[HH + c + 3] + e2 * sWe[2*HH + c + 3] + e3 * sWe[3*HH + c + 3] + sbe[c + 3];
    atomicAdd(ap + 0, fmaxf(m0, 0.f));
    atomicAdd(ap + 1, fmaxf(m1, 0.f));
    atomicAdd(ap + 2, fmaxf(m2, 0.f));
    atomicAdd(ap + 3, fmaxf(m3, 0.f));
}

// ---------------------------------------------------------------------------
// Fused MLP + LayerNorm + relu + residual (per layer), WMMA bf16.
// Block = 256 threads = 8 waves; 32-node tile (two 16-row M-tiles sharing
// every B fragment -> each weight load feeds two v_wmma).  In-place on h.
// ---------------------------------------------------------------------------
__global__ void __launch_bounds__(256)
k_mlp_ln(const float* __restrict__ agg, const __bf16* __restrict__ W1t,
         const float* __restrict__ b1, const __bf16* __restrict__ W2t,
         const float* __restrict__ b2, const float* __restrict__ gam,
         const float* __restrict__ bet, float* __restrict__ h, int l) {
    __shared__ __bf16 Z[32 * ZS];   // z tile, bf16
    __shared__ __bf16 U[32 * US];   // hidden relu(zW1+b1), bf16
    __shared__ float  Y[32 * YS];   // output of second GEMM, f32
    __shared__ float  MU[32], RS[32];

    const int tid = threadIdx.x;
    const int lane = tid & 31, wave = tid >> 5;
    const int lane15 = lane & 15;
    const int hl = lane >> 4;                 // which 16-lane half
    const size_t node0 = (size_t)blockIdx.x * 32;

    // ---- stage 0: z = h + agg  ->  bf16 LDS (32 rows) ----
    for (int hf = 0; hf < 2; ++hf) {
        int row = (tid >> 4) + hf * 16;       // 0..31
        int cb = (tid & 15) * 8;              // 0..120
        // NOTE: last block's rows >= NN read into the adjacent agg region
        // (allocated workspace) and are never written back.
        const float* hp = h + (node0 + row) * HH + cb;
        const float* ap = agg + (node0 + row) * HH + cb;
        float4 h0 = *(const float4*)hp,       h1 = *(const float4*)(hp + 4);
        float4 a0 = *(const float4*)ap,       a1 = *(const float4*)(ap + 4);
        __bf16* zp = &Z[row * ZS + cb];
        zp[0] = (__bf16)(h0.x + a0.x); zp[1] = (__bf16)(h0.y + a0.y);
        zp[2] = (__bf16)(h0.z + a0.z); zp[3] = (__bf16)(h0.w + a0.w);
        zp[4] = (__bf16)(h1.x + a1.x); zp[5] = (__bf16)(h1.y + a1.y);
        zp[6] = (__bf16)(h1.z + a1.z); zp[7] = (__bf16)(h1.w + a1.w);
    }
    __syncthreads();

    const __bf16* W1l = W1t + (size_t)l * 2 * HH * HH;
    const __bf16* W2l = W2t + (size_t)l * HH * 2 * HH;

    // ---- stage 1: U = relu(Z @ W1 + b1), Z:[32,128] W1:[128,256] ----
    for (int t = 0; t < 2; ++t) {
        int n0 = (wave * 2 + t) * 16;
        v8f acc0 = {}, acc1 = {};
        for (int k0 = 0; k0 < HH; k0 += 32) {
            // B fragment: col N = n0+lane15; K = k0 + {0..7,16..23} (+8 for upper lane half) (shared)
            const __bf16* wb = W1l + ((size_t)(n0 + lane15)) * HH + k0 + hl * 8;
            bf16x8 blo = *(const bf16x8*)wb;
            bf16x8 bhi = *(const bf16x8*)(wb + 16);
            bf16x16 b;
            for (int i = 0; i < 8; ++i) { b[i] = blo[i]; b[8 + i] = bhi[i]; }
            // A fragments: row M = lane15 (m-tile 0) / 16+lane15 (m-tile 1);
            // K = k0 + {0..7,16..23} (+8 for upper lane half)
            const __bf16* za0 = &Z[lane15 * ZS + k0 + hl * 8];
            const __bf16* za1 = &Z[(16 + lane15) * ZS + k0 + hl * 8];
            bf16x8 alo0 = *(const bf16x8*)za0, ahi0 = *(const bf16x8*)(za0 + 16);
            bf16x8 alo1 = *(const bf16x8*)za1, ahi1 = *(const bf16x8*)(za1 + 16);
            bf16x16 a0, a1;
            for (int i = 0; i < 8; ++i) {
                a0[i] = alo0[i]; a0[8 + i] = ahi0[i];
                a1[i] = alo1[i]; a1[8 + i] = ahi1[i];
            }
            acc0 = __builtin_amdgcn_wmma_f32_16x16x32_bf16(false, a0, false, b,
                                                           (short)0, acc0, false, false);
            acc1 = __builtin_amdgcn_wmma_f32_16x16x32_bf16(false, a1, false, b,
                                                           (short)0, acc1, false, false);
        }
        float bias = b1[l * 2 * HH + n0 + lane15];
        for (int r = 0; r < 8; ++r) {
            float v0 = fmaxf(acc0[r] + bias, 0.f);
            float v1 = fmaxf(acc1[r] + bias, 0.f);
            U[(r + 8 * hl) * US + n0 + lane15]        = (__bf16)v0;
            U[(16 + r + 8 * hl) * US + n0 + lane15]   = (__bf16)v1;
        }
    }
    __syncthreads();

    // ---- stage 2: Y = U @ W2 + b2, U:[32,256] W2:[256,128] ----
    {
        int n0 = wave * 16;
        v8f acc0 = {}, acc1 = {};
        for (int k0 = 0; k0 < 2 * HH; k0 += 32) {
            const __bf16* wb = W2l + ((size_t)(n0 + lane15)) * 2 * HH + k0 + hl * 8;
            bf16x8 blo = *(const bf16x8*)wb;
            bf16x8 bhi = *(const bf16x8*)(wb + 16);
            bf16x16 b;
            for (int i = 0; i < 8; ++i) { b[i] = blo[i]; b[8 + i] = bhi[i]; }
            const __bf16* ua0 = &U[lane15 * US + k0 + hl * 8];
            const __bf16* ua1 = &U[(16 + lane15) * US + k0 + hl * 8];
            bf16x8 alo0 = *(const bf16x8*)ua0, ahi0 = *(const bf16x8*)(ua0 + 16);
            bf16x8 alo1 = *(const bf16x8*)ua1, ahi1 = *(const bf16x8*)(ua1 + 16);
            bf16x16 a0, a1;
            for (int i = 0; i < 8; ++i) {
                a0[i] = alo0[i]; a0[8 + i] = ahi0[i];
                a1[i] = alo1[i]; a1[8 + i] = ahi1[i];
            }
            acc0 = __builtin_amdgcn_wmma_f32_16x16x32_bf16(false, a0, false, b,
                                                           (short)0, acc0, false, false);
            acc1 = __builtin_amdgcn_wmma_f32_16x16x32_bf16(false, a1, false, b,
                                                           (short)0, acc1, false, false);
        }
        float bias = b2[l * HH + n0 + lane15];
        for (int r = 0; r < 8; ++r) {
            Y[(r + 8 * hl) * YS + n0 + lane15]      = acc0[r] + bias;
            Y[(16 + r + 8 * hl) * YS + n0 + lane15] = acc1[r] + bias;
        }
    }
    __syncthreads();

    // ---- LayerNorm stats: wave w reduces rows {4w+hl, 4w+2+hl} ----
    for (int rr = 0; rr < 2; ++rr) {
        int row = wave * 4 + rr * 2 + hl;
        const float* yp = &Y[row * YS + lane15 * 8];
        float s = 0.f, s2 = 0.f;
        for (int i = 0; i < 8; ++i) { float v = yp[i]; s += v; s2 += v * v; }
        for (int off = 1; off < 16; off <<= 1) {
            s  += __shfl_xor(s,  off, 32);
            s2 += __shfl_xor(s2, off, 32);
        }
        if (lane15 == 0) {
            float mu = s * (1.f / HH);
            float var = s2 * (1.f / HH) - mu * mu;
            MU[row] = mu;
            RS[row] = rsqrtf(var + 1e-5f);
        }
    }
    __syncthreads();

    // ---- normalize + relu + residual, write h in place (guarded tail) ----
    for (int hf = 0; hf < 2; ++hf) {
        int row = (tid >> 4) + hf * 16;
        size_t node = node0 + row;
        if (node >= NN) break;
        int cb = (tid & 15) * 8;
        float mu = MU[row], rs = RS[row];
        float* hp = h + node * HH + cb;
        const float* gp = gam + l * HH + cb;
        const float* bp = bet + l * HH + cb;
        for (int i = 0; i < 8; ++i) {
            float v = (Y[row * YS + cb + i] - mu) * rs * gp[i] + bp[i];
            hp[i] += fmaxf(v, 0.f);
        }
    }
}

// ---------------------------------------------------------------------------
// global_add_pool: out[batch[n]] += h[n].  One wave per node.
// ---------------------------------------------------------------------------
__global__ void k_pool(const float* __restrict__ h, const int* __restrict__ batch,
                       float* __restrict__ out) {
    int wave = threadIdx.x >> 5, lane = threadIdx.x & 31;
    int n = blockIdx.x * 8 + wave;
    if (n >= NN) return;
    int g = batch[n];
    int c = lane * 4;
    float4 hv = *(const float4*)(h + (size_t)n * HH + c);
    float* op = out + (size_t)g * HH + c;
    atomicAdd(op + 0, hv.x);
    atomicAdd(op + 1, hv.y);
    atomicAdd(op + 2, hv.z);
    atomicAdd(op + 3, hv.w);
}

extern "C" void kernel_launch(void* const* d_in, const int* in_sizes, int n_in,
                              void* d_out, int out_size, void* d_ws, size_t ws_size,
                              hipStream_t stream) {
    (void)in_sizes; (void)n_in; (void)out_size; (void)ws_size;
    const float* x     = (const float*)d_in[0];
    const int*   ei    = (const int*)d_in[1];
    const float* eattr = (const float*)d_in[2];
    const int*   batch = (const int*)d_in[3];
    const float* Wn    = (const float*)d_in[4];
    const float* bn    = (const float*)d_in[5];
    const float* We    = (const float*)d_in[6];
    const float* be    = (const float*)d_in[7];
    const float* W1    = (const float*)d_in[8];
    const float* b1    = (const float*)d_in[9];
    const float* W2    = (const float*)d_in[10];
    const float* b2    = (const float*)d_in[11];
    const float* gam   = (const float*)d_in[12];
    const float* bet   = (const float*)d_in[13];
    float* out = (float*)d_out;

    char* ws = (char*)d_ws;
    float*  h   = (float*)(ws);                         // 25,600,000 B
    float*  agg = (float*)(ws + 25600000);              // 25,600,000 B
    __bf16* W1t = (__bf16*)(ws + 51200000);             //    262,144 B
    __bf16* W2t = (__bf16*)(ws + 51462144);             //    262,144 B

    k_wconv<<<512, 256, 0, stream>>>(W1, W2, W1t, W2t);
    k_encode<<<6250, 256, 0, stream>>>(x, Wn, bn, h);
    for (int l = 0; l < LL; ++l) {
        k_zero<<<6250, 256, 0, stream>>>(agg, NN * HH);
        k_msg_agg<<<78125, 256, 0, stream>>>(h, ei, eattr, We, be, agg);
        k_mlp_ln<<<1563, 256, 0, stream>>>(agg, W1t, b1, W2t, b2, gam, bet, h, l);
    }
    k_zero<<<250, 256, 0, stream>>>(out, GG * HH);
    k_pool<<<6250, 256, 0, stream>>>(h, batch, out);
}
